// GraphEncoder_49340584297066
// MI455X (gfx1250) — matrix operand order repaired and checked by reading, not compiled, not run on hardware
//
#include <hip/hip_runtime.h>
#include <hip/hip_bf16.h>

typedef __attribute__((ext_vector_type(16))) _Float16 v16h;
typedef __attribute__((ext_vector_type(8)))  _Float16 v8h;
typedef __attribute__((ext_vector_type(8)))  float    v8f;
typedef __attribute__((ext_vector_type(4)))  unsigned u32x4;
typedef __attribute__((ext_vector_type(8)))  int      i32x8;
typedef __attribute__((ext_vector_type(4)))  int      i32x4;

#define DEV __device__ __forceinline__

constexpr int B_ = 16, N_ = 1000, D_ = 128, H_ = 8, DH = 16, F_ = 512;
constexpr int NTQ  = 63;         // query tiles of 16 (covers 1008 >= 1000)
constexpr int NTK  = 64;         // key/value tiles of 16 (1024, for 32-key chunks)
constexpr int NPAD = NTK * 16;   // 1024
constexpr int NC   = NPAD / 32;  // 32 key chunks of 32

DEV v8f wmma16(v16h a, v16h b, v8f c) {
  // D = A(16x32 f16) * B(32x16 f16) + C(16x16 f32)
  return __builtin_amdgcn_wmma_f32_16x16x32_f16(false, a, false, b, (short)0, c,
                                                false, false);
}

// A-matrix loader (16x32 f16). lane L: row = L&15; element j maps to
// K = kbase + (j<8 ? 0 : 16) + ((L>=16)?8:0) + (j&7)  -> two contiguous 16B loads.
DEV v16h load_a(const _Float16* __restrict__ rowp, int kbase, int lane) {
  const int h8off = (lane >> 4) * 8;
  union { v16h v; v8h h[2]; } u;
  u.h[0] = *(const v8h*)(rowp + kbase + h8off);
  u.h[1] = *(const v8h*)(rowp + kbase + 16 + h8off);
  return u.v;
}

// B-matrix loader (32x16 f16) from a transposed (N x K) row-major source.
// lane L: column n = L&15; element j maps to
// K = kbase + (j<8 ? 0 : 16) + ((L>=16)?8:0) + (j&7)
DEV v16h load_b(const _Float16* __restrict__ baseT, int ld, int kbase, int lane) {
  const int n   = lane & 15;
  const int h8off = (lane >> 4) * 8;
  union { v16h v; v8h h[2]; } u;
  u.h[0] = *(const v8h*)(baseT + (size_t)n * ld + kbase + h8off);
  u.h[1] = *(const v8h*)(baseT + (size_t)n * ld + kbase + 16 + h8off);
  return u.v;
}

// Tensor Data Mover: copy an (h x w) f16 tile (row stride = `stride` elements)
// from global memory into LDS at byte offset lds_off. D# packed per ISA 8.3/8.4.
DEV void tdm_copy(unsigned lds_off, const _Float16* g, unsigned w, unsigned h,
                  unsigned stride) {
  const unsigned long long ga = (unsigned long long)(const void*)g;
  u32x4 g0;
  g0[0] = 1u;                                     // count=1, user mode
  g0[1] = lds_off;                                // lds_addr
  g0[2] = (unsigned)ga;                           // global_addr[31:0]
  g0[3] = ((unsigned)(ga >> 32) & 0x01ffffffu)    // global_addr[56:32]
          | 0x80000000u;                          // type=2 ("image")
  i32x8 g1;
  g1[0] = 0x00010000;                             // data_size=1 -> 2 bytes
  g1[1] = (int)((stride & 0xffffu) << 16);        // tensor_dim0[15:0]
  g1[2] = (int)((stride >> 16) & 0xffffu)         // tensor_dim0[31:16]
          | (int)((h & 0xffffu) << 16);           // tensor_dim1[15:0]
  g1[3] = (int)((w & 0xffffu) << 16);             // tile_dim0
  g1[4] = (int)(h & 0xffffu);                     // tile_dim1 (tile_dim2=0)
  g1[5] = (int)stride;                            // tensor_dim0_stride[31:0]
  g1[6] = 0;
  g1[7] = 0;
  const i32x4 z4 = {0, 0, 0, 0};
  const i32x8 z8 = {0, 0, 0, 0, 0, 0, 0, 0};
  __builtin_amdgcn_tensor_load_to_lds(g0, g1, z4, z4, z8, 0);
}

// ---------------------------------------------------------------- prep kernels
__global__ void k_cvt_x(const float* __restrict__ x, _Float16* __restrict__ xh, int n) {
  for (int i = blockIdx.x * blockDim.x + threadIdx.x; i < n;
       i += gridDim.x * blockDim.x)
    xh[i] = (_Float16)x[i];
}

__global__ void k_prep_w(const float* __restrict__ wq, const float* __restrict__ wk,
                         const float* __restrict__ wv, const float* __restrict__ wo,
                         const float* __restrict__ w1, const float* __restrict__ w2,
                         _Float16* __restrict__ wqt, _Float16* __restrict__ wkt,
                         _Float16* __restrict__ wvt, _Float16* __restrict__ wot,
                         _Float16* __restrict__ w1t, _Float16* __restrict__ w2t) {
  const int i = blockIdx.x * blockDim.x + threadIdx.x;
  const int stride = gridDim.x * blockDim.x;
  for (int t = i; t < H_ * DH * D_; t += stride) {   // W{q,k,v}: (H,D,Dh)->(H,Dh,D)
    const int d = t % D_; const int dh = (t / D_) % DH; const int h = t / (D_ * DH);
    const int s = h * D_ * DH + d * DH + dh;
    wqt[t] = (_Float16)wq[s];
    wkt[t] = (_Float16)wk[s];
    wvt[t] = (_Float16)wv[s];
  }
  for (int t = i; t < D_ * D_; t += stride) {        // Wo: (H,Dh,D)->(D,H*Dh)
    const int hd = t % D_; const int d = t / D_;
    wot[t] = (_Float16)wo[hd * D_ + d];
  }
  for (int t = i; t < F_ * D_; t += stride) {        // W1: (D,F)->(F,D)
    const int d = t % D_; const int f = t / D_;
    w1t[t] = (_Float16)w1[d * F_ + f];
  }
  for (int t = i; t < D_ * F_; t += stride) {        // W2: (F,D)->(D,F)
    const int f = t % F_; const int d = t / F_;
    w2t[t] = (_Float16)w2[f * D_ + d];
  }
}

// ---------------------------------------------------------------- QKV projection
__global__ __launch_bounds__(32)
void k_qkv(const _Float16* __restrict__ xh, const _Float16* __restrict__ wqt,
           const _Float16* __restrict__ wkt, const _Float16* __restrict__ wvt,
           _Float16* __restrict__ Qh, _Float16* __restrict__ Kh,
           _Float16* __restrict__ Vt) {
  const int lane = threadIdx.x & 31;
  int idx = blockIdx.x;
  const int qt = idx % NTK; idx /= NTK;
  const int b  = idx % B_;  idx /= B_;
  const int h  = idx;
  const int tok0 = qt * 16;
  const int r = lane & 15;
  int tokA = tok0 + r; if (tokA >= N_) tokA = N_ - 1;   // clamp padded rows
  const _Float16* xrow = xh + ((size_t)b * N_ + tokA) * D_;

  v16h a[4];
#pragma unroll
  for (int k = 0; k < 4; ++k) a[k] = load_a(xrow, k * 32, lane);

  const _Float16* wq = wqt + (size_t)h * DH * D_;
  const _Float16* wk = wkt + (size_t)h * DH * D_;
  const _Float16* wv = wvt + (size_t)h * DH * D_;

  v8f cq = {}, ck = {}, cv = {};
#pragma unroll
  for (int k = 0; k < 4; ++k) {
    cq = wmma16(a[k], load_b(wq, D_, k * 32, lane), cq);
    ck = wmma16(a[k], load_b(wk, D_, k * 32, lane), ck);
    cv = wmma16(a[k], load_b(wv, D_, k * 32, lane), cv);
  }

  const int col   = lane & 15;
  const int h8off = (lane >> 4) * 8;
  const size_t hb = (size_t)h * B_ + b;
  _Float16* qb = Qh + hb * NPAD * DH;
  _Float16* kb = Kh + hb * NPAD * DH;
  _Float16* vb = Vt + hb * DH * NPAD;       // V stored transposed (Dh x NPAD)
#pragma unroll
  for (int v = 0; v < 8; ++v) {
    const int tok = tok0 + v + h8off;
    const bool ok = tok < N_;
    qb[(size_t)tok * DH + col]   = (_Float16)(cq[v] * 0.25f);  // fold 1/sqrt(Dh)
    kb[(size_t)tok * DH + col]   = ok ? (_Float16)ck[v] : (_Float16)0.0f;
    vb[(size_t)col * NPAD + tok] = ok ? (_Float16)cv[v] : (_Float16)0.0f;
  }
}

// ------------------------------------------- flash attention, TDM double-buffered
__global__ __launch_bounds__(32)
void k_attn(const _Float16* __restrict__ Qh, const _Float16* __restrict__ Kh,
            const _Float16* __restrict__ Vt, _Float16* __restrict__ ctx16) {
  __shared__ alignas(64) _Float16 kbuf[2][32 * DH];   // 32 keys x 16 dh
  __shared__ alignas(64) _Float16 vbuf[2][DH * 32];   // 16 dh  x 32 keys
  __shared__ alignas(64) _Float16 plds[16 * 32];      // P tile transpose staging
  const int lane = threadIdx.x & 31;
  int idx = blockIdx.x;
  const int qt = idx % NTQ; idx /= NTQ;
  const int b  = idx % B_;  idx /= B_;
  const int h  = idx;
  const size_t hb = (size_t)h * B_ + b;
  const _Float16* qb = Qh + hb * NPAD * DH;
  const _Float16* kb = Kh + hb * NPAD * DH;
  const _Float16* vb = Vt + hb * DH * NPAD;

  const int r     = lane & 15;
  const int col   = lane & 15;
  const int h8off = (lane >> 4) * 8;
  const int k16   = (lane >> 4) * 16;
  v8h zz = {};

  // Q tile as A-matrix 16x32 with K=16..31 zero (Dh=16 padded)
  const _Float16* qrow = qb + (size_t)(qt * 16 + r) * DH;
  __builtin_prefetch(qrow, 0, 1);
  union { v16h v; v8h h[2]; } uq;
  uq.h[0] = lane < 16 ? *(const v8h*)(qrow) : zz;
  uq.h[1] = lane < 16 ? *(const v8h*)(qrow + 8) : zz;

  float m[8], l[8];
  v8f o = {};
#pragma unroll
  for (int v = 0; v < 8; ++v) { m[v] = -3.0e38f; l[v] = 0.0f; }

  // prologue: DMA chunk 0 into buffer 0
  tdm_copy((unsigned)(size_t)&kbuf[0][0], kb, 32 * DH, 1, 32 * DH);
  tdm_copy((unsigned)(size_t)&vbuf[0][0], vb, 32, DH, NPAD);

  for (int c = 0; c < NC; ++c) {
    const int cur = c & 1;
    if (c + 1 < NC) {   // overlap next chunk's DMA with this chunk's math
      tdm_copy((unsigned)(size_t)&kbuf[cur ^ 1][0],
               kb + (size_t)(c + 1) * 32 * DH, 32 * DH, 1, 32 * DH);
      tdm_copy((unsigned)(size_t)&vbuf[cur ^ 1][0],
               vb + (size_t)(c + 1) * 32, 32, DH, NPAD);
      __builtin_amdgcn_s_wait_tensorcnt(2);   // chunk c's two loads complete
    } else {
      __builtin_amdgcn_s_wait_tensorcnt(0);
    }
    asm volatile("" ::: "memory");

    const _Float16* kc = kbuf[cur];
    const _Float16* vc = vbuf[cur];

    // S tiles for 32 keys: B[k][n] = K[key n][dh k], k>=16 zero (Dh pad)
    v16h bk0 = {}, bk1 = {};
    if (lane < 16) {
      bk0 = *(const v16h*)(kc + col * DH);
      bk1 = *(const v16h*)(kc + (16 + col) * DH);
    }
    v8f zc = {};
    v8f s0 = wmma16(uq.v, bk0, zc);
    v8f s1 = wmma16(uq.v, bk1, zc);

    const int  key0 = c * 32 + col;
    const int  key1 = key0 + 16;
    const bool ok0 = key0 < N_, ok1 = key1 < N_;
    float p0[8], p1[8];
#pragma unroll
    for (int v = 0; v < 8; ++v) {
      const float a0 = ok0 ? s0[v] : -3.0e38f;
      const float a1 = ok1 ? s1[v] : -3.0e38f;
      float mv = fmaxf(a0, a1);
#pragma unroll
      for (int off = 1; off < 16; off <<= 1)
        mv = fmaxf(mv, __shfl_xor(mv, off, 16));   // row max within lane-half
      const float mnew = fmaxf(m[v], mv);
      const float sc = __expf(m[v] - mnew);
      const float e0 = __expf(a0 - mnew);
      const float e1 = __expf(a1 - mnew);
      float rs = e0 + e1;
#pragma unroll
      for (int off = 1; off < 16; off <<= 1)
        rs += __shfl_xor(rs, off, 16);             // row sum
      l[v] = l[v] * sc + rs;
      m[v] = mnew;
      o[v] *= sc;
      p0[v] = e0; p1[v] = e1;
    }

    // D-layout P (16x32) -> LDS -> A-layout P
#pragma unroll
    for (int v = 0; v < 8; ++v) {
      plds[(v + h8off) * 32 + col]      = (_Float16)p0[v];
      plds[(v + h8off) * 32 + 16 + col] = (_Float16)p1[v];
    }
    asm volatile("s_wait_dscnt 0x0" ::: "memory");
    union { v16h v; v8h h[2]; } up;
    up.h[0] = *(const v8h*)(plds + r * 32 + k16);
    up.h[1] = *(const v8h*)(plds + r * 32 + k16 + 8);
    asm volatile("" ::: "memory");

    // O += P * V over all 32 keys: B[k][n] = V[key k][dh n] (dense, all lanes)
    v16h bv = *(const v16h*)(vc + col * 32 + k16);
    o = wmma16(up.v, bv, o);
  }

  _Float16* cb = ctx16 + (size_t)b * NPAD * D_ + h * DH;
#pragma unroll
  for (int v = 0; v < 8; ++v) {
    const int tok = qt * 16 + v + h8off;
    if (tok < N_)
      cb[(size_t)tok * D_ + col] = (_Float16)(o[v] / l[v]);
  }
}

// ---------------------------------------------------------------- out-proj + residual
__global__ __launch_bounds__(32)
void k_proj(const _Float16* __restrict__ ctx16, const _Float16* __restrict__ wot,
            const float* __restrict__ x, float* __restrict__ hpre) {
  const int lane = threadIdx.x & 31;
  int idx = blockIdx.x;
  const int qt = idx % NTQ; idx /= NTQ;
  const int b  = idx;
  const int tok0 = qt * 16;
  const int r = lane & 15;
  const _Float16* crow = ctx16 + ((size_t)b * NPAD + tok0 + r) * D_;
  v16h a[4];
#pragma unroll
  for (int k = 0; k < 4; ++k) a[k] = load_a(crow, k * 32, lane);

  const int col   = lane & 15;
  const int h8off = (lane >> 4) * 8;
#pragma unroll
  for (int nt = 0; nt < 8; ++nt) {
    v8f c = {};
#pragma unroll
    for (int k = 0; k < 4; ++k)
      c = wmma16(a[k], load_b(wot + (size_t)nt * 16 * D_, D_, k * 32, lane), c);
    const int d = nt * 16 + col;
#pragma unroll
    for (int v = 0; v < 8; ++v) {
      const int tok = tok0 + v + h8off;
      if (tok < N_) {
        const size_t off = ((size_t)b * N_ + tok) * D_ + d;
        hpre[off] = c[v] + x[off];
      }
    }
  }
}

// ---------------------------------------------------------------- joint LayerNorm
__global__ __launch_bounds__(256)
void k_ln(const float* __restrict__ src, float* __restrict__ dstf,
          _Float16* __restrict__ dsth) {
  __shared__ float s0[256], s1[256];
  const int b = blockIdx.x;
  const int tid = threadIdx.x;
  const size_t base = (size_t)b * N_ * D_;
  const int NE = N_ * D_;
  float sum = 0.f, sq = 0.f;
  for (int i = tid; i < NE; i += 256) {
    const float v = src[base + i];
    sum += v; sq += v * v;
  }
  s0[tid] = sum; s1[tid] = sq;
  __syncthreads();
  for (int w = 128; w > 0; w >>= 1) {
    if (tid < w) { s0[tid] += s0[tid + w]; s1[tid] += s1[tid + w]; }
    __syncthreads();
  }
  const float mean = s0[0] / (float)NE;
  const float var  = (s1[0] - (float)NE * mean * mean) / (float)(NE - 1); // ddof=1
  const float rstd = rsqrtf(var + 1e-5f);
  for (int i = tid; i < NE; i += 256) {
    const float v = (src[base + i] - mean) * rstd;
    dstf[base + i] = v;
    if (dsth) {
      const int row = i / D_, cc = i % D_;
      dsth[((size_t)b * NPAD + row) * D_ + cc] = (_Float16)v;
    }
  }
}

// ---------------------------------------------------------------- feed-forward
__global__ __launch_bounds__(32)
void k_ff(const _Float16* __restrict__ h16, const float* __restrict__ hf,
          const _Float16* __restrict__ w1t, const float* __restrict__ b1,
          const _Float16* __restrict__ w2t, const float* __restrict__ b2,
          float* __restrict__ ffpre) {
  __shared__ alignas(64) _Float16 t16[16 * F_];   // 16KB activation tile in LDS
  const int lane = threadIdx.x & 31;
  int idx = blockIdx.x;
  const int qt = idx % NTQ; idx /= NTQ;
  const int b  = idx;
  const int tok0  = qt * 16;
  const int r     = lane & 15;
  const int col   = lane & 15;
  const int h8off = (lane >> 4) * 8;

  const _Float16* hrow = h16 + ((size_t)b * NPAD + tok0 + r) * D_;
  v16h a[4];
#pragma unroll
  for (int k = 0; k < 4; ++k) a[k] = load_a(hrow, k * 32, lane);

  // FF1: t = relu(h @ W1 + b1)  (32 column tiles of F)
  for (int nt = 0; nt < 32; ++nt) {
    const int f = nt * 16 + col;
    const float bias = b1[f];
    v8f c = { bias, bias, bias, bias, bias, bias, bias, bias };
#pragma unroll
    for (int k = 0; k < 4; ++k)
      c = wmma16(a[k], load_b(w1t + (size_t)nt * 16 * D_, D_, k * 32, lane), c);
#pragma unroll
    for (int v = 0; v < 8; ++v)
      t16[(v + h8off) * F_ + f] = (_Float16)fmaxf(c[v], 0.0f);
  }
  asm volatile("s_wait_dscnt 0x0" ::: "memory");

  // FF2: y = t @ W2 + b2 + h   (8 column tiles of D, K=512 contraction)
#pragma unroll
  for (int nt = 0; nt < 8; ++nt) {
    const int d = nt * 16 + col;
    const float bias = b2[d];
    v8f c = { bias, bias, bias, bias, bias, bias, bias, bias };
    for (int k = 0; k < 16; ++k) {
      union { v16h v; v8h h[2]; } ua;
      ua.h[0] = *(const v8h*)(t16 + r * F_ + k * 32 + h8off);
      ua.h[1] = *(const v8h*)(t16 + r * F_ + k * 32 + 16 + h8off);
      c = wmma16(ua.v, load_b(w2t + (size_t)nt * 16 * F_, F_, k * 32, lane), c);
    }
#pragma unroll
    for (int v = 0; v < 8; ++v) {
      const int tok = tok0 + v + h8off;
      if (tok < N_) {
        const size_t off = ((size_t)b * N_ + tok) * D_ + d;
        ffpre[off] = c[v] + hf[off];
      }
    }
  }
}

// ---------------------------------------------------------------- launch
extern "C" void kernel_launch(void* const* d_in, const int* in_sizes, int n_in,
                              void* d_out, int out_size, void* d_ws, size_t ws_size,
                              hipStream_t stream) {
  const float* x  = (const float*)d_in[0];
  const float* Wq = (const float*)d_in[1];
  const float* Wk = (const float*)d_in[2];
  const float* Wv = (const float*)d_in[3];
  const float* Wo = (const float*)d_in[4];
  const float* W1 = (const float*)d_in[5];
  const float* b1 = (const float*)d_in[6];
  const float* W2 = (const float*)d_in[7];
  const float* b2 = (const float*)d_in[8];
  float* out = (float*)d_out;

  char* ws = (char*)d_ws;
  size_t off = 0;
  auto alloc = [&](size_t bytes) -> void* {
    void* p = ws + off;
    off += (bytes + 255) & ~(size_t)255;
    return p;
  };
  _Float16* xh    = (_Float16*)alloc((size_t)B_ * N_ * D_ * 2);
  _Float16* wqt   = (_Float16*)alloc((size_t)H_ * DH * D_ * 2);
  _Float16* wkt   = (_Float16*)alloc((size_t)H_ * DH * D_ * 2);
  _Float16* wvt   = (_Float16*)alloc((size_t)H_ * DH * D_ * 2);
  _Float16* wot   = (_Float16*)alloc((size_t)D_ * D_ * 2);
  _Float16* w1t   = (_Float16*)alloc((size_t)F_ * D_ * 2);
  _Float16* w2t   = (_Float16*)alloc((size_t)D_ * F_ * 2);
  _Float16* Qh    = (_Float16*)alloc((size_t)H_ * B_ * NPAD * DH * 2);
  _Float16* Kh    = (_Float16*)alloc((size_t)H_ * B_ * NPAD * DH * 2);
  _Float16* Vt    = (_Float16*)alloc((size_t)H_ * B_ * DH * NPAD * 2);
  _Float16* ctx16 = (_Float16*)alloc((size_t)B_ * NPAD * D_ * 2);
  float*    hpre  = (float*)   alloc((size_t)B_ * N_ * D_ * 4);
  float*    hbuf  = (float*)   alloc((size_t)B_ * N_ * D_ * 4);
  _Float16* h16   = (_Float16*)alloc((size_t)B_ * NPAD * D_ * 2);
  float*    ffpre = (float*)   alloc((size_t)B_ * N_ * D_ * 4);
  (void)ws_size; (void)in_sizes; (void)n_in; (void)out_size;

  k_cvt_x<<<512, 256, 0, stream>>>(x, xh, B_ * N_ * D_);
  k_prep_w<<<64, 256, 0, stream>>>(Wq, Wk, Wv, Wo, W1, W2,
                                   wqt, wkt, wvt, wot, w1t, w2t);
  k_qkv<<<H_ * B_ * NTK, 32, 0, stream>>>(xh, wqt, wkt, wvt, Qh, Kh, Vt);
  k_attn<<<H_ * B_ * NTQ, 32, 0, stream>>>(Qh, Kh, Vt, ctx16);
  k_proj<<<B_ * NTQ, 32, 0, stream>>>(ctx16, wot, x, hpre);
  k_ln<<<B_, 256, 0, stream>>>(hpre, hbuf, h16);
  k_ff<<<B_ * NTQ, 32, 0, stream>>>(h16, hbuf, w1t, b1, w2t, b2, ffpre);
  k_ln<<<B_, 256, 0, stream>>>(ffpre, out, (_Float16*)nullptr);
}
